// ContrastiveScorer_6914897347195
// MI455X (gfx1250) — matrix operand order repaired and checked by reading, not compiled, not run on hardware
//
#include <hip/hip_runtime.h>

#define HEADS 4
#define HD 64
#define HW 256   // HEADS*HD

typedef __bf16 bf16x16 __attribute__((ext_vector_type(16)));
typedef float  f32x8   __attribute__((ext_vector_type(8)));

enum { ACT_NONE = 0, ACT_ELU = 1, ACT_LRELU = 2, ACT_RELU = 3 };

__device__ __forceinline__ unsigned short f2bf(float f) {
    // round-to-nearest-even fp32 -> bf16
    unsigned int u = __float_as_uint(f);
    u += 0x7FFFu + ((u >> 16) & 1u);
    return (unsigned short)(u >> 16);
}

__device__ __forceinline__ float apply_act(float v, int act) {
    switch (act) {
        case ACT_ELU:   return v > 0.f ? v : (__expf(v) - 1.f);
        case ACT_LRELU: return v > 0.f ? v : 0.01f * v;
        case ACT_RELU:  return v > 0.f ? v : 0.f;
        default:        return v;
    }
}

// fp32 -> bf16(as ushort) elementwise
__global__ void cvt_bf16_kernel(const float* __restrict__ in,
                                unsigned short* __restrict__ out, int n) {
    int t = blockIdx.x * blockDim.x + threadIdx.x;
    if (t < n) out[t] = f2bf(in[t]);
}

// out[row, outoff+col] = act(A[nrows,K] @ W[K,M] + bias).
// A, W pre-converted to bf16 (ushort storage). K%32==0, M%64==0.
// Block = 256 thr = 8 waves; block tile = 128 rows x 64 cols.
// Wave tile = 16 rows x 64 cols: one A fragment reused across 4 B fragments
// (4 v_wmma_f32_16x16x32_bf16 per k-step). W tile staged in LDS transposed
// ([col][K]) so each B fragment is two contiguous ds_load_b128.
// A row index is clamped instead of masked: out-of-range rows compute garbage
// that is discarded by the store-side guard, keeping the hot loop branch-free.
__global__ __launch_bounds__(256)
void wmma_gemm_kernel(const unsigned short* __restrict__ A,
                      const unsigned short* __restrict__ W,
                      const float* __restrict__ bias, float* __restrict__ out,
                      int nrows, int K, int M, int ldout, int outoff, int act)
{
    __shared__ __align__(16) unsigned short smem[64 * 256];  // 32 KB, [col 0..63][K]

    const int tid  = threadIdx.x;
    const int wave = tid >> 5;
    const int lane = tid & 31;
    const int l16  = lane & 15;
    const int hi   = lane >> 4;                 // half-wave select
    const int row0 = blockIdx.x * 128 + wave * 16;
    const int col0 = blockIdx.y * 64;

    // ---- stage W[:, col0:col0+64] into LDS, transposed
    const int chunks = (K * 64) / 8;            // 8 bf16 per 16B chunk
    for (int idx = tid; idx < chunks; idx += 256) {
        const int k  = idx >> 3;
        const int c8 = (idx & 7) * 8;
        union { uint4 q; unsigned short s[8]; } u;
        u.q = *(const uint4*)(W + (size_t)k * M + col0 + c8);
        #pragma unroll
        for (int j = 0; j < 8; ++j) smem[(c8 + j) * K + k] = u.s[j];
    }
    __syncthreads();

    const int arow = min(row0 + l16, nrows - 1);       // clamped, not masked
    const unsigned short* arp = A + (size_t)arow * K;

    f32x8 acc[4] = {{}, {}, {}, {}};
    union Frag { bf16x16 v; uint4 q[2]; };

    for (int k0 = 0; k0 < K; k0 += 32) {
        // A fragment: elems 0..7 <-> K = k0+8*hi .. +7 ; elems 8..15 <-> +16 more
        Frag af;
        af.q[0] = *(const uint4*)(arp + k0 + hi * 8);
        af.q[1] = *(const uint4*)(arp + k0 + hi * 8 + 16);

        const int kb = k0 + hi * 8;             // B elems 0..7 <-> K = kb + e ; elems 8..15 <-> +16 more
        #pragma unroll
        for (int cc = 0; cc < 4; ++cc) {
            Frag bf;
            const unsigned short* wp = smem + (cc * 16 + l16) * K + kb;
            bf.q[0] = *(const uint4*)(wp);
            bf.q[1] = *(const uint4*)(wp + 16);
            acc[cc] = __builtin_amdgcn_wmma_f32_16x16x32_bf16(false, af.v, false, bf.v,
                                                              (short)0, acc[cc], false, false);
        }
    }

    #pragma unroll
    for (int cc = 0; cc < 4; ++cc) {
        const int ocol = col0 + cc * 16 + l16;
        const float bv = bias ? bias[ocol] : 0.f;
        #pragma unroll
        for (int r = 0; r < 8; ++r) {
            const int orow = row0 + r + hi * 8;
            if (orow < nrows)
                out[(size_t)orow * ldout + outoff + ocol] = apply_act(acc[cc][r] + bv, act);
        }
    }
}

__global__ void fill_kernel(float* __restrict__ p, float v, int n) {
    int t = blockIdx.x * blockDim.x + threadIdx.x;
    if (t < n) p[t] = v;
}

// e_src[n,h] = <h[n,h,:], a_s[h,:]>, e_dst likewise.
__global__ void score_kernel(const float* __restrict__ h, const float* __restrict__ a_s,
                             const float* __restrict__ a_d,
                             float* __restrict__ esrc, float* __restrict__ edst, int n)
{
    int t = blockIdx.x * blockDim.x + threadIdx.x;
    if (t >= n * HEADS) return;
    const int node = t >> 2, hd = t & 3;
    const float* hp = h   + (size_t)node * HW + hd * HD;
    const float* ap = a_s + hd * HD;
    const float* bp = a_d + hd * HD;
    float ss = 0.f, sd = 0.f;
    #pragma unroll 4
    for (int k = 0; k < HD; ++k) { float hv = hp[k]; ss += hv * ap[k]; sd += hv * bp[k]; }
    esrc[t] = ss; edst[t] = sd;
}

__device__ __forceinline__ void edge_sd(const int* __restrict__ ei, int E, int t,
                                        int* s, int* d) {
    if (t < E) { *s = ei[t]; *d = ei[E + t]; }
    else       { *s = t - E; *d = t - E; }       // appended self-loops
}

__device__ __forceinline__ float edge_e(const float* __restrict__ esrc,
                                        const float* __restrict__ edst,
                                        int s, int d, int hd) {
    float e = esrc[s * HEADS + hd] + edst[d * HEADS + hd];
    return e > 0.f ? e : 0.2f * e;               // GAT leaky relu slope
}

__device__ __forceinline__ void atomicMaxFloat(float* addr, float val) {
    unsigned int* ua = (unsigned int*)addr;
    unsigned int old = *ua;
    while (__uint_as_float(old) < val) {
        unsigned int assumed = old;
        old = atomicCAS(ua, assumed, __float_as_uint(val));
        if (old == assumed) break;
    }
}

__global__ void edge_max_kernel(const int* __restrict__ ei, int E, int n,
                                const float* __restrict__ esrc, const float* __restrict__ edst,
                                float* __restrict__ m)
{
    int t = blockIdx.x * blockDim.x + threadIdx.x;
    if (t >= (E + n) * HEADS) return;
    int s, d; edge_sd(ei, E, t >> 2, &s, &d);
    const int hd = t & 3;
    atomicMaxFloat(&m[d * HEADS + hd], edge_e(esrc, edst, s, d, hd));
}

__global__ void edge_sum_kernel(const int* __restrict__ ei, int E, int n,
                                const float* __restrict__ esrc, const float* __restrict__ edst,
                                const float* __restrict__ m, float* __restrict__ ssum)
{
    int t = blockIdx.x * blockDim.x + threadIdx.x;
    if (t >= (E + n) * HEADS) return;
    int s, d; edge_sd(ei, E, t >> 2, &s, &d);
    const int hd = t & 3;
    const int di = d * HEADS + hd;
    atomicAdd(&ssum[di], __expf(edge_e(esrc, edst, s, d, hd) - m[di]));
}

__global__ void edge_agg_kernel(const int* __restrict__ ei, int E, int n,
                                const float* __restrict__ esrc, const float* __restrict__ edst,
                                const float* __restrict__ m, const float* __restrict__ ssum,
                                const float* __restrict__ h, float* __restrict__ agg)
{
    int t = blockIdx.x * blockDim.x + threadIdx.x;
    if (t >= (E + n) * HEADS) return;
    int s, d; edge_sd(ei, E, t >> 2, &s, &d);
    const int hd = t & 3;
    const int di = d * HEADS + hd;
    const float alpha = __expf(edge_e(esrc, edst, s, d, hd) - m[di]) / (ssum[di] + 1e-16f);
    const float* hp = h   + (size_t)s * HW + hd * HD;
    float*       op = agg + (size_t)d * HW + hd * HD;
    #pragma unroll 4
    for (int k = 0; k < HD; ++k) atomicAdd(&op[k], alpha * hp[k]);
}

// concat=true path: out[n, 0:256] = ELU(agg + bias)
__global__ void finalize_concat_kernel(const float* __restrict__ agg, const float* __restrict__ b,
                                       float* __restrict__ out, int n)
{
    int t = blockIdx.x * blockDim.x + threadIdx.x;
    if (t >= n * HW) return;
    out[t] = apply_act(agg[t] + b[t % HW], ACT_ELU);
}

// concat=false path: out[n, 0:64] = LeakyReLU_0.01(mean over heads + bias)
__global__ void finalize_mean_kernel(const float* __restrict__ agg, const float* __restrict__ b,
                                     float* __restrict__ out, int n)
{
    int t = blockIdx.x * blockDim.x + threadIdx.x;
    if (t >= n * HD) return;
    const int node = t / HD, dcol = t - node * HD;
    const float* ap = agg + (size_t)node * HW + dcol;
    float v = 0.25f * (ap[0] + ap[HD] + ap[2 * HD] + ap[3 * HD]) + b[dcol];
    out[t] = apply_act(v, ACT_LRELU);
}

// out[n] = sigmoid(<hid[n,:64], w> + b)
__global__ void final_head_kernel(const float* __restrict__ hid, const float* __restrict__ w,
                                  const float* __restrict__ b, float* __restrict__ out, int n)
{
    int t = blockIdx.x * blockDim.x + threadIdx.x;
    if (t >= n) return;
    const float* hp = hid + (size_t)t * HD;
    float s = b[0];
    #pragma unroll 4
    for (int k = 0; k < HD; ++k) s += hp[k] * w[k];
    out[t] = 1.f / (1.f + __expf(-s));
}

// ------------------------- host orchestration -------------------------

struct Ctx {
    int N, E;
    float *h, *agg, *esrc, *edst, *m, *s;       // fp32 scratch
    unsigned short *abf, *wbf;                  // bf16 scratch
    hipStream_t stream;
};

static inline int cdiv(int a, int b) { return (a + b - 1) / b; }

// out = act(A @ W + bias), fp32 in/out, bf16 tensor cores inside.
static void gemm_bf16(const Ctx& c, const float* A, const float* W, const float* bias,
                      float* out, int nrows, int K, int M, int ldout, int outoff, int act)
{
    const int na = nrows * K, nw = K * M;
    cvt_bf16_kernel<<<cdiv(na, 256), 256, 0, c.stream>>>(A, c.abf, na);
    cvt_bf16_kernel<<<cdiv(nw, 256), 256, 0, c.stream>>>(W, c.wbf, nw);
    dim3 g(cdiv(nrows, 128), M / 64);
    wmma_gemm_kernel<<<g, 256, 0, c.stream>>>(c.abf, c.wbf, bias, out,
                                              nrows, K, M, ldout, outoff, act);
}

static void run_gatconv(const Ctx& c, const int* ei, const float* x, int Kin,
                        const float* W, const float* as_, const float* ad_, const float* b,
                        bool concat, float* out)
{
    const int N = c.N, E = c.E;
    gemm_bf16(c, x, W, nullptr, c.h, N, Kin, HW, HW, 0, ACT_NONE);   // h = x @ W

    const int nh = N * HEADS;
    score_kernel<<<cdiv(nh, 256), 256, 0, c.stream>>>(c.h, as_, ad_, c.esrc, c.edst, N);
    fill_kernel<<<cdiv(nh, 256), 256, 0, c.stream>>>(c.m, -1e30f, nh);
    fill_kernel<<<cdiv(nh, 256), 256, 0, c.stream>>>(c.s, 0.f, nh);
    const int na = N * HW;
    fill_kernel<<<cdiv(na, 256), 256, 0, c.stream>>>(c.agg, 0.f, na);

    const int tot = (E + N) * HEADS;
    edge_max_kernel<<<cdiv(tot, 256), 256, 0, c.stream>>>(ei, E, N, c.esrc, c.edst, c.m);
    edge_sum_kernel<<<cdiv(tot, 256), 256, 0, c.stream>>>(ei, E, N, c.esrc, c.edst, c.m, c.s);
    edge_agg_kernel<<<cdiv(tot, 256), 256, 0, c.stream>>>(ei, E, N, c.esrc, c.edst,
                                                          c.m, c.s, c.h, c.agg);
    if (concat)
        finalize_concat_kernel<<<cdiv(na, 256), 256, 0, c.stream>>>(c.agg, b, out, N);
    else
        finalize_mean_kernel<<<cdiv(N * HD, 256), 256, 0, c.stream>>>(c.agg, b, out, N);
}

// one GAT block: conv1 (concat, ELU fused) -> conv2 (mean, trailing LeakyReLU fused)
static void run_gat(const Ctx& c, const int* ei, const float* x, int Kin,
                    const float* const* p, float* mid, float* out)
{
    run_gatconv(c, ei, x,   Kin, p[0], p[1], p[2], p[3], true,  mid);
    run_gatconv(c, ei, mid, HW,  p[4], p[5], p[6], p[7], false, out);
}

extern "C" void kernel_launch(void* const* d_in, const int* in_sizes, int n_in,
                              void* d_out, int out_size, void* d_ws, size_t ws_size,
                              hipStream_t stream)
{
    (void)n_in; (void)out_size; (void)ws_size;
    const int N = in_sizes[0] / 128;          // 50000
    const int E = in_sizes[2] / 2;            // 800000

    const float* x[2]  = { (const float*)d_in[0], (const float*)d_in[1] };
    const int*   ei[2] = { (const int*)d_in[2],   (const int*)d_in[3]   };

    const float* g0[8] = { (const float*)d_in[4],  (const float*)d_in[5],
                           (const float*)d_in[6],  (const float*)d_in[7],
                           (const float*)d_in[8],  (const float*)d_in[9],
                           (const float*)d_in[10], (const float*)d_in[11] };
    const float* g1[8] = { (const float*)d_in[12], (const float*)d_in[13],
                           (const float*)d_in[14], (const float*)d_in[15],
                           (const float*)d_in[16], (const float*)d_in[17],
                           (const float*)d_in[18], (const float*)d_in[19] };
    const float* enc_W = (const float*)d_in[20];
    const float* enc_b = (const float*)d_in[21];
    const float* h0_W  = (const float*)d_in[22];
    const float* h0_b  = (const float*)d_in[23];
    const float* h1_W  = (const float*)d_in[24];
    const float* h1_b  = (const float*)d_in[25];

    // workspace carve-out
    float* ws   = (float*)d_ws;
    float* B0   = ws;                            // [N,256] conv1 (concat) output
    float* B1   = B0 + (size_t)N * HW;           // [N,256] h = x@W
    float* B2   = B1 + (size_t)N * HW;           // [N,256] softmax aggregate
    float* B3   = B2 + (size_t)N * HW;           // [N,64]  per-GAT output
    float* esrc = B3 + (size_t)N * HD;           // [N,4]
    float* edst = esrc + (size_t)N * HEADS;
    float* mbuf = edst + (size_t)N * HEADS;
    float* sbuf = mbuf + (size_t)N * HEADS;
    float* zbuf = sbuf + (size_t)N * HEADS;      // [N,128] = [z1 | z2]
    unsigned short* abf = (unsigned short*)(zbuf + (size_t)N * 2 * HD); // [N,256] bf16
    unsigned short* wbf = abf + (size_t)N * HW;  // [256,256] bf16
    float* hid  = B1;                            // [N,64], safe reuse after encodes

    Ctx c{ N, E, B1, B2, esrc, edst, mbuf, sbuf, abf, wbf, stream };

    for (int g = 0; g < 2; ++g) {
        run_gat(c, ei[g], x[g], 128, g0, B0, B3);   // block 0
        run_gat(c, ei[g], B3,   HD,  g1, B0, B3);   // shared block, pass 1
        run_gat(c, ei[g], B3,   HD,  g1, B0, B3);   // shared block, pass 2
        // z_g = B3 @ enc_W + enc_b, into zbuf[:, g*64 : g*64+64]
        gemm_bf16(c, B3, enc_W, enc_b, zbuf, N, HD, HD, 2 * HD, g * HD, ACT_NONE);
    }

    // contrastive head: hid = relu(z @ h0_W + h0_b); out = sigmoid(hid @ h1_W + h1_b)
    gemm_bf16(c, zbuf, h0_W, h0_b, hid, N, 2 * HD, HD, HD, 0, ACT_RELU);
    final_head_kernel<<<cdiv(N, 256), 256, 0, stream>>>(hid, h1_W, h1_b, (float*)d_out, N);
}
